// Model_39676907886903
// MI455X (gfx1250) — matrix operand order repaired and checked by reading, not compiled, not run on hardware
//
#include <hip/hip_runtime.h>
#include <hip/hip_bf16.h>

typedef __attribute__((ext_vector_type(16))) _Float16 v16h;
typedef __attribute__((ext_vector_type(4)))  _Float16 v4h;
typedef __attribute__((ext_vector_type(8)))  float    v8f;

#define S_LEN   4096
#define D_LEN   64
#define KC      32
#define NITER   (S_LEN / KC)
#define HALF_N  33554432u   /* (4*4096*4096)/2 : threefry pair offset        */
#define B_STRIDE 16777216u  /* 4096*4096       : flat batch stride           */

__device__ __forceinline__ uint32_t rotl32(uint32_t x, uint32_t n) {
  return (x << n) | (x >> (32u - n));
}

// JAX threefry2x32 with key(42) -> key data (0, 42).
__device__ __forceinline__ void threefry2x32(uint32_t x0, uint32_t x1,
                                             uint32_t &o0, uint32_t &o1) {
  const uint32_t k0 = 0u, k1 = 42u;
  const uint32_t k2 = k0 ^ k1 ^ 0x1BD11BDAu;
  x0 += k0; x1 += k1;
#define TF_R(r) { x0 += x1; x1 = rotl32(x1, r); x1 ^= x0; }
  TF_R(13u) TF_R(15u) TF_R(26u) TF_R(6u)
  x0 += k1; x1 += k2 + 1u;
  TF_R(17u) TF_R(29u) TF_R(16u) TF_R(24u)
  x0 += k2; x1 += k0 + 2u;
  TF_R(13u) TF_R(15u) TF_R(26u) TF_R(6u)
  x0 += k0; x1 += k1 + 3u;
  TF_R(17u) TF_R(29u) TF_R(16u) TF_R(24u)
  x0 += k1; x1 += k2 + 4u;
  TF_R(13u) TF_R(15u) TF_R(26u) TF_R(6u)
  x0 += k2; x1 += k0 + 5u;
#undef TF_R
  o0 = x0; o1 = x1;
}

// bits -> uniform[-0.1, 0.1) (JAX mantissa trick) -> exp(u)
__device__ __forceinline__ float tf_uniform_exp(uint32_t bits) {
  float u = __uint_as_float((bits >> 9) | 0x3F800000u) - 1.0f;
  float v = fmaxf(-0.1f, u * 0.2f - 0.1f);
  return __expf(v);
}

// One block = one 16-row s-tile for batch pair {b0, b0+2}.
// 4 waves, each owns a 16-wide d-slice; 2 WMMA accumulators per wave (one per batch).
__global__ __launch_bounds__(128) void fused_rand_softmax_av(
    const float* __restrict__ input, float* __restrict__ out) {
  __shared__ _Float16 Pt[2][16][KC];       // prob tiles, 2 KB
  __shared__ _Float16 Bt[2][D_LEN][40];    // input chunk, t-minor, padded rows (10 KB)
  __shared__ float    rs[2][16];           // running row sums of p

  const int tid   = threadIdx.x;
  const int lane  = tid & 31;
  const int wave  = tid >> 5;
  const int b0    = blockIdx.x & 1;        // handles batches b0 and b0+2
  const int sTile = blockIdx.x >> 1;       // 0..255
  const int sBase = sTile * 16;

  if (tid < 32) rs[tid >> 4][tid & 15] = 0.0f;
  __syncthreads();

  v8f acc0 = {};
  v8f acc1 = {};

  // P-generation assignment: 16 rows x 32 cols, 4 consecutive cols per thread
  const int pr  = tid >> 3;                // row 0..15
  const int pcb = (tid & 7) * 4;           // col base
  const uint32_t jrow =
      (uint32_t)b0 * B_STRIDE + (uint32_t)(sBase + pr) * (uint32_t)S_LEN;

  // WMMA fragment geometry (16-bit A 16x32 / B 32x16 layouts, wave32)
  const int hi    = (lane >> 4) & 1;
  const int l15   = lane & 15;
  const int aKoff = hi ? 16 : 0;           // A: lanes16-31 hold K 16..31
  const int bKoff = hi ? 16 : 0;           // B: lanes16-31 hold K 16..31
  const int dcol  = wave * 16 + l15;       // this lane's output column d

  const float4* __restrict__ in4 = (const float4*)input;

  for (int kc = 0; kc < NITER; ++kc) {
    const int t0 = kc * KC;

    // ---- stage input[b, t0:t0+32, :] for both batches into LDS (f32->f16),
    //      d-major / t-minor so B fragments are contiguous 32B reads.
    //      8 x float4 (global_load_b128) per thread per iteration.
    #pragma unroll
    for (int q = 0; q < 8; ++q) {
      int idx = q * 128 + tid;             // 0..1023 float4 units
      int bb  = idx >> 9;                  // 0: batch b0, 1: batch b0+2
      int rem = idx & 511;
      int tt  = rem >> 4;                  // t within chunk, 0..31
      int dd4 = rem & 15;                  // float4 block along d
      int bsel = b0 + 2 * bb;
      float4 vl = in4[((size_t)bsel * S_LEN + (size_t)(t0 + tt)) * 16 + dd4];
      int dd = dd4 * 4;
      Bt[bb][dd + 0][tt] = (_Float16)vl.x;
      Bt[bb][dd + 1][tt] = (_Float16)vl.y;
      Bt[bb][dd + 2][tt] = (_Float16)vl.z;
      Bt[bb][dd + 3][tt] = (_Float16)vl.w;
    }

    // ---- generate p = exp(uniform) for both batches; one threefry per pair
    float sum0 = 0.f, sum1 = 0.f;
    v4h pk0, pk1;
    #pragma unroll
    for (int k = 0; k < 4; ++k) {
      uint32_t j = jrow + (uint32_t)(t0 + pcb + k);
      uint32_t o0, o1;
      threefry2x32(j, j + HALF_N, o0, o1);
      float p0 = tf_uniform_exp(o0);       // batch b0
      float p1 = tf_uniform_exp(o1);       // batch b0+2
      pk0[k] = (_Float16)p0;
      pk1[k] = (_Float16)p1;
      sum0 += p0; sum1 += p1;
    }
    *(v4h*)&Pt[0][pr][pcb] = pk0;          // one ds_store_b64 each
    *(v4h*)&Pt[1][pr][pcb] = pk1;
    atomicAdd(&rs[0][pr], sum0);
    atomicAdd(&rs[1][pr], sum1);
    __syncthreads();

    // ---- load fragments from LDS in contiguous k order (k = 16*hi + e) for both A and B
    v16h a0, a1, f0, f1;
    {
      const _Float16* r0 = &Pt[0][l15][0];
      const _Float16* r1 = &Pt[1][l15][0];
      #pragma unroll
      for (int e = 0; e < 16; ++e) {
        a0[e]     = r0[aKoff + e];
        a1[e]     = r1[aKoff + e];
      }
      const _Float16* c0 = &Bt[0][dcol][0];
      const _Float16* c1 = &Bt[1][dcol][0];
      #pragma unroll
      for (int e = 0; e < 16; ++e) {
        f0[e] = c0[bKoff + e];
        f1[e] = c1[bKoff + e];
      }
    }

    acc0 = __builtin_amdgcn_wmma_f32_16x16x32_f16(false, a0, false, f0,
                                                  (short)0, acc0, false, false);
    acc1 = __builtin_amdgcn_wmma_f32_16x16x32_f16(false, a1, false, f1,
                                                  (short)0, acc1, false, false);
    __syncthreads();   // protect LDS tiles before next iteration overwrites
  }

  // ---- normalize: softmax denom = rowsum; column-L1 denom = B = 4 exactly.
  // denom ~ 4*4096, so a 1-ulp v_rcp_f32 is ample precision.
  #pragma unroll
  for (int v = 0; v < 8; ++v) {
    int m    = v + (hi ? 8 : 0);           // C/D layout: lanes16-31 hold M 8..15
    int srow = sBase + m;
    float inv0 = __builtin_amdgcn_rcpf(4.0f * rs[0][m]);
    float inv1 = __builtin_amdgcn_rcpf(4.0f * rs[1][m]);
    out[((size_t)b0 * S_LEN + srow) * D_LEN + dcol]       = acc0[v] * inv0;
    out[((size_t)(b0 + 2) * S_LEN + srow) * D_LEN + dcol] = acc1[v] * inv1;
  }
}

extern "C" void kernel_launch(void* const* d_in, const int* in_sizes, int n_in,
                              void* d_out, int out_size, void* d_ws, size_t ws_size,
                              hipStream_t stream) {
  (void)in_sizes; (void)n_in; (void)out_size; (void)d_ws; (void)ws_size;
  const float* input = (const float*)d_in[0];   // [4,4096,64] f32
  // d_in[1] (attn_mask) is dead in the reference: all-False and the masked
  // score tensor is overwritten by fresh uniform weights.
  float* out = (float*)d_out;                   // [4,4096,64] f32

  dim3 grid(512);   // 256 s-tiles * 2 batch pairs {0,2} / {1,3}
  dim3 block(128);  // 4 waves
  hipLaunchKernelGGL(fused_rand_softmax_av, grid, block, 0, stream, input, out);
}
